// HDMI_90572270338206
// MI455X (gfx1250) — matrix operand order repaired and checked by reading, not compiled, not run on hardware
//
#include <hip/hip_runtime.h>
#include <hip/hip_bf16.h>
#include <math.h>

// Problem constants (from reference)
#define NN    50000
#define DINN  512
#define DOUTN 128

// Affine pseudo-permutation (gcd(7919, 50000) = 1)
#define PERM_A 7919
#define PERM_B 12345

// W^T staged as bf16 with padded row stride (512 + 8) to avoid LDS bank conflicts
#define WT_STRIDE 520

typedef __attribute__((ext_vector_type(16))) __bf16 v16bf_t;
typedef __attribute__((ext_vector_type(8)))  __bf16 v8bf_t;
typedef __attribute__((ext_vector_type(8)))  float  v8f_t;
typedef __attribute__((ext_vector_type(4)))  float  v4f_t;

// ---------------------------------------------------------------------------
// Deterministic integer hash (stand-in for JAX threefry; no runtime check)
// ---------------------------------------------------------------------------
__device__ __forceinline__ unsigned dgi_hash(unsigned x) {
    x ^= x >> 16; x *= 0x7feb352dU;
    x ^= x >> 15; x *= 0x846ca68bU;
    x ^= x >> 16;
    return x;
}

// ---------------------------------------------------------------------------
// Dropout: produce pos_seq and neg_seq (both are reference outputs)
// ---------------------------------------------------------------------------
__global__ __launch_bounds__(256)
void dgi_dropout(const float* __restrict__ x,
                 float* __restrict__ pos_seq,
                 float* __restrict__ neg_seq,
                 const float* __restrict__ p_ptr,
                 unsigned total)
{
    unsigned i = blockIdx.x * 256u + threadIdx.x;
    if (i >= total) return;
    float p = *p_ptr;
    float keep = 1.0f - p;
    float scale = 1.0f / fmaxf(keep, 1e-12f);
    unsigned thresh = (unsigned)(fminf(fmaxf(keep, 0.0f), 1.0f) * 4294967040.0f);
    float xi = x[i];
    unsigned h1 = dgi_hash(i ^ 0x12345678u);
    unsigned h2 = dgi_hash(i ^ 0x9e3779b9u);
    pos_seq[i] = (h1 < thresh) ? xi * scale : 0.0f;
    neg_seq[i] = (h2 < thresh) ? xi * scale : 0.0f;
}

// ---------------------------------------------------------------------------
// Zero the pos_h/neg_h accumulation region of d_out
// ---------------------------------------------------------------------------
__global__ __launch_bounds__(256)
void dgi_zero(float* __restrict__ dst, unsigned total)
{
    unsigned i = blockIdx.x * 256u + threadIdx.x;
    if (i < total) dst[i] = 0.0f;
}

// deg[i] = 1.0 (self loop)
__global__ __launch_bounds__(256)
void dgi_init_deg(float* __restrict__ deg)
{
    unsigned i = blockIdx.x * 256u + threadIdx.x;
    if (i < NN) deg[i] = 1.0f;
}

// deg[dst] += 1 per edge
__global__ __launch_bounds__(256)
void dgi_deg_accum(const int* __restrict__ dst_idx, float* __restrict__ deg, int E)
{
    int e = blockIdx.x * 256 + threadIdx.x;
    if (e < E) atomicAdd(&deg[dst_idx[e]], 1.0f);
}

// dinv = rsqrt(deg) in place (deg >= 1 always)
__global__ __launch_bounds__(256)
void dgi_dinv(float* __restrict__ deg)
{
    unsigned i = blockIdx.x * 256u + threadIdx.x;
    if (i < NN) deg[i] = rsqrtf(deg[i]);
}

// ---------------------------------------------------------------------------
// Convert W [512,128] f32 -> Wt [128][WT_STRIDE] bf16 (transposed, padded)
// ---------------------------------------------------------------------------
__global__ __launch_bounds__(256)
void dgi_convert_w(const float* __restrict__ W, __bf16* __restrict__ Wt)
{
    unsigned i = blockIdx.x * 256u + threadIdx.x;
    if (i >= DINN * DOUTN) return;
    unsigned k = i >> 7;      // 0..511
    unsigned n = i & 127;     // 0..127
    Wt[(size_t)n * WT_STRIDE + k] = (__bf16)W[i];
}

// ---------------------------------------------------------------------------
// GEMM: hw[N,128] = seq(perm?)[N,512] @ W[512,128] via v_wmma_f32_16x16x32_bf16
// One wave per 16-row M-tile, all 8 N-tiles (16x128 block), K in chunks of 32.
// Wt (bf16, transposed) is staged into 130 KB of dynamic LDS per workgroup via
// the CDNA5 async-to-LDS datapath (GLOBAL_LOAD_ASYNC_TO_LDS_B128 / ASYNCcnt),
// which bypasses VGPRs entirely.
// ---------------------------------------------------------------------------
__global__ __launch_bounds__(256)
void dgi_gemm_wmma(const float* __restrict__ seq,
                   const __bf16* __restrict__ Wt,
                   float* __restrict__ hw,
                   int usePerm)
{
    extern __shared__ unsigned char lds_raw[];
    __bf16* ldsW = (__bf16*)lds_raw;

    // Async copy of Wt into LDS: 128*520*2 = 133120 B = 8320 x b128.
    // Only dynamic LDS is used by this kernel, so the dynamic region starts at
    // LDS offset 0; VDST carries the per-lane LDS byte offset (HW adds LDS_BASE).
    {
        const char* g = (const char*)Wt;
        for (int i = threadIdx.x; i < (DOUTN * WT_STRIDE * 2) / 16; i += 256) {
            unsigned lds_off = (unsigned)i * 16u;
            const char* gaddr = g + (size_t)i * 16u;
            asm volatile("global_load_async_to_lds_b128 %0, %1, off"
                         :: "v"(lds_off), "v"(gaddr)
                         : "memory");
        }
        asm volatile("s_wait_asynccnt 0x0" ::: "memory");
    }
    __syncthreads();

    const int wave  = threadIdx.x >> 5;
    const int lane  = threadIdx.x & 31;
    const int mtile = blockIdx.x * 8 + wave;
    if (mtile >= NN / 16) return;

    const int half = lane >> 4;   // 0: lanes 0-15, 1: lanes 16-31
    const int mr   = lane & 15;

    int m = mtile * 16 + mr;
    int row = usePerm ? ((m * PERM_A + PERM_B) % NN) : m;
    const float* arow = seq + (size_t)row * DINN;

    v8f_t acc[8] = {};   // 16x128 f32 accumulators

    for (int kc = 0; kc < DINN / 32; ++kc) {
        // A fragment (16x32 bf16) per CDNA5 ISA layout:
        // lanes 0-15: elems 0-7 -> K=kc*32+0..7,  elems 8-15 -> K=kc*32+16..23
        // lanes 16-31: elems 0-7 -> K=kc*32+8..15, elems 8-15 -> K=kc*32+24..31
        const int k0 = kc * 32 + half * 8;
        __builtin_prefetch(arow + k0 + 128, 0, 1);
        v4f_t a0 = *(const v4f_t*)(arow + k0);
        v4f_t a1 = *(const v4f_t*)(arow + k0 + 4);
        v4f_t a2 = *(const v4f_t*)(arow + k0 + 16);
        v4f_t a3 = *(const v4f_t*)(arow + k0 + 20);
        v16bf_t afrag;
        afrag[0]  = (__bf16)a0.x; afrag[1]  = (__bf16)a0.y;
        afrag[2]  = (__bf16)a0.z; afrag[3]  = (__bf16)a0.w;
        afrag[4]  = (__bf16)a1.x; afrag[5]  = (__bf16)a1.y;
        afrag[6]  = (__bf16)a1.z; afrag[7]  = (__bf16)a1.w;
        afrag[8]  = (__bf16)a2.x; afrag[9]  = (__bf16)a2.y;
        afrag[10] = (__bf16)a2.z; afrag[11] = (__bf16)a2.w;
        afrag[12] = (__bf16)a3.x; afrag[13] = (__bf16)a3.y;
        afrag[14] = (__bf16)a3.z; afrag[15] = (__bf16)a3.w;

        // B fragment (32x16 bf16): lane n = tile_n*16 + mr,
        // same K order as A: elems 0-7 -> K = kb..kb+7, elems 8-15 -> K = kb+16..kb+23
        const int kb = kc * 32 + half * 8;
        #pragma unroll
        for (int nt = 0; nt < 8; ++nt) {
            const __bf16* bp = ldsW + (size_t)(nt * 16 + mr) * WT_STRIDE + kb;
            v8bf_t b0 = *(const v8bf_t*)bp;        // ds_load_b128
            v8bf_t b1 = *(const v8bf_t*)(bp + 16); // ds_load_b128
            v16bf_t bfrag = __builtin_shufflevector(
                b0, b1, 0, 1, 2, 3, 4, 5, 6, 7, 8, 9, 10, 11, 12, 13, 14, 15);
            acc[nt] = __builtin_amdgcn_wmma_f32_16x16x32_bf16(
                /*neg_a=*/false, afrag, /*neg_b=*/false, bfrag,
                /*c_mod=*/(short)0, acc[nt],
                /*reuse_a=*/false, /*reuse_b=*/false);
        }
    }

    // C/D layout: VGPR r -> M = r (lanes 0-15) / 8+r (lanes 16-31), N = lane&15
    const int mbase = mtile * 16 + half * 8;
    #pragma unroll
    for (int nt = 0; nt < 8; ++nt) {
        const int col = nt * 16 + mr;
        #pragma unroll
        for (int r = 0; r < 8; ++r)
            hw[(size_t)(mbase + r) * DOUTN + col] = acc[nt][r];
    }
}

// ---------------------------------------------------------------------------
// Edge scatter: one wave per edge, each lane owns 4 of the 128 columns.
// out[dst] += hw[src] * dinv[src]*dinv[dst]  (global_atomic_add_f32)
// ---------------------------------------------------------------------------
__global__ __launch_bounds__(256)
void dgi_scatter(const int* __restrict__ src_idx,
                 const int* __restrict__ dst_idx,
                 const float* __restrict__ hw,
                 const float* __restrict__ dinv,
                 float* __restrict__ out,
                 int E)
{
    const int lane   = threadIdx.x & 31;
    const int wid    = (blockIdx.x * 256 + threadIdx.x) >> 5;
    const int nwaves = (gridDim.x * 256) >> 5;
    for (int e = wid; e < E; e += nwaves) {
        int s = src_idx[e];
        int d = dst_idx[e];
        float norm = dinv[s] * dinv[d];
        v4f_t v = ((const v4f_t*)(hw + (size_t)s * DOUTN))[lane];
        float* op = out + (size_t)d * DOUTN + lane * 4;
        atomicAdd(op + 0, v.x * norm);
        atomicAdd(op + 1, v.y * norm);
        atomicAdd(op + 2, v.z * norm);
        atomicAdd(op + 3, v.w * norm);
    }
}

// ---------------------------------------------------------------------------
// Self-loop contribution + bias + ReLU (in place over out region)
// ---------------------------------------------------------------------------
__global__ __launch_bounds__(256)
void dgi_finalize(float* __restrict__ out,
                  const float* __restrict__ hw,
                  const float* __restrict__ dinv,
                  const float* __restrict__ bias)
{
    unsigned idx = blockIdx.x * 256u + threadIdx.x;
    if (idx >= (unsigned)NN * DOUTN) return;
    unsigned i = idx >> 7;
    unsigned c = idx & 127;
    float di = dinv[i];
    float v = out[idx] + hw[idx] * di * di + bias[c];
    out[idx] = fmaxf(v, 0.0f);
}

// ---------------------------------------------------------------------------
// summary[c] = sigmoid(mean_i pos_h[i,c]); one block per column
// ---------------------------------------------------------------------------
__global__ __launch_bounds__(256)
void dgi_summary(const float* __restrict__ pos_h, float* __restrict__ summ)
{
    __shared__ float red[256];
    const int c = blockIdx.x;
    float s = 0.0f;
    for (int i = threadIdx.x; i < NN; i += 256)
        s += pos_h[(size_t)i * DOUTN + c];
    red[threadIdx.x] = s;
    __syncthreads();
    for (int off = 128; off > 0; off >>= 1) {
        if (threadIdx.x < (unsigned)off) red[threadIdx.x] += red[threadIdx.x + off];
        __syncthreads();
    }
    if (threadIdx.x == 0) {
        float m = red[0] * (1.0f / (float)NN);
        summ[c] = 1.0f / (1.0f + __expf(-m));
    }
}

// ---------------------------------------------------------------------------
// Host-side orchestration
// ---------------------------------------------------------------------------
extern "C" void kernel_launch(void* const* d_in, const int* in_sizes, int n_in,
                              void* d_out, int out_size, void* d_ws, size_t ws_size,
                              hipStream_t stream)
{
    const float* x    = (const float*)d_in[0];   // [N,512]
    const int*   ei   = (const int*)d_in[1];     // [2,E] flat
    const float* W    = (const float*)d_in[2];   // [512,128]
    const float* bias = (const float*)d_in[3];   // [128]
    const float* p    = (const float*)d_in[4];   // scalar

    const int E = in_sizes[1] / 2;
    const int* src_idx = ei;       // edge_index[0] (gather side)
    const int* dst_idx = ei + E;   // edge_index[1] (scatter side)

    // Output layout (tuple flattened in return order)
    float* out      = (float*)d_out;
    float* pos_h    = out;                            // 6,400,000
    float* neg_h    = out + (size_t)NN * DOUTN;       // 6,400,000
    float* summ     = neg_h + (size_t)NN * DOUTN;     // 128
    float* pos_seq  = summ + DOUTN;                   // 25,600,000
    float* neg_seq  = pos_seq + (size_t)NN * DINN;    // 25,600,000

    // Workspace layout
    float*  hw_pos = (float*)d_ws;
    float*  hw_neg = hw_pos + (size_t)NN * DOUTN;
    float*  dinv   = hw_neg + (size_t)NN * DOUTN;          // deg -> dinv in place
    __bf16* Wt     = (__bf16*)(dinv + NN);                 // 16B aligned

    const unsigned totSeq = (unsigned)NN * DINN;   // 25,600,000
    const unsigned totH   = (unsigned)NN * DOUTN;  //  6,400,000

    // 1) Dropout -> pos_seq, neg_seq
    dgi_dropout<<<(totSeq + 255) / 256, 256, 0, stream>>>(x, pos_seq, neg_seq, p, totSeq);

    // 2) Zero pos_h + neg_h accumulators (contiguous region)
    dgi_zero<<<(2 * totH + 255) / 256, 256, 0, stream>>>(pos_h, 2 * totH);

    // 3) Degrees (self-loop init, edge accumulate, rsqrt)
    dgi_init_deg<<<(NN + 255) / 256, 256, 0, stream>>>(dinv);
    dgi_deg_accum<<<(E + 255) / 256, 256, 0, stream>>>(dst_idx, dinv, E);
    dgi_dinv<<<(NN + 255) / 256, 256, 0, stream>>>(dinv);

    // 4) W -> bf16 transposed staging buffer
    dgi_convert_w<<<(DINN * DOUTN + 255) / 256, 256, 0, stream>>>(W, Wt);

    // 5) WMMA GEMMs: hw = seq @ W (neg branch gathers rows via affine perm)
    const int gemmBlocks = ((NN / 16) + 7) / 8;                 // 391
    const size_t ldsBytes = (size_t)DOUTN * WT_STRIDE * 2;      // 133,120 B
    dgi_gemm_wmma<<<gemmBlocks, 256, ldsBytes, stream>>>(pos_seq, Wt, hw_pos, 0);
    dgi_gemm_wmma<<<gemmBlocks, 256, ldsBytes, stream>>>(neg_seq, Wt, hw_neg, 1);

    // 6) Edge scatter-add (wave per edge, atomics mostly L2-resident)
    dgi_scatter<<<2048, 256, 0, stream>>>(src_idx, dst_idx, hw_pos, dinv, pos_h, E);
    dgi_scatter<<<2048, 256, 0, stream>>>(src_idx, dst_idx, hw_neg, dinv, neg_h, E);

    // 7) Self-loop + bias + ReLU
    dgi_finalize<<<(totH + 255) / 256, 256, 0, stream>>>(pos_h, hw_pos, dinv, bias);
    dgi_finalize<<<(totH + 255) / 256, 256, 0, stream>>>(neg_h, hw_neg, dinv, bias);

    // 8) summary = sigmoid(mean(pos_h, axis=0))
    dgi_summary<<<DOUTN, 256, 0, stream>>>(pos_h, summ);
}
